// SonataCodec_38062000177394
// MI455X (gfx1250) — matrix operand order repaired and checked by reading, not compiled, not run on hardware
//
#include <hip/hip_runtime.h>
#include <hip/hip_bf16.h>
#include <math.h>

// ---------------------------------------------------------------------------
// SonataCodec forward on MI455X (gfx1250).
// bf16 WMMA implicit-GEMM convs + RVQ GEMMs, f32 epilogues, wave32.
// Each wave computes a 32x32 output tile = 4 x v_wmma_f32_16x16x32_bf16 per
// 32-wide K chunk. LDS tiles are kept in *fragment order* so fragment loads
// are two ds_load_b128 per matrix; staging uses unconditional clamped loads
// (no EXEC divergence) and packed bf16x2 LDS stores.
// ---------------------------------------------------------------------------

typedef __attribute__((ext_vector_type(16))) __bf16 v16bf;
typedef __attribute__((ext_vector_type(2)))  __bf16 v2bf;
typedef __attribute__((ext_vector_type(8)))  float  v8f;

#define BATCH   64
#define TLAT    249
#define BT      15936      // 64 * 249
#define DLAT    512
#define DPROJ   128
#define NCODES  1024
#define NBOOKS  8

__device__ __forceinline__ __bf16 to_bf(float f) { return (__bf16)f; }

__device__ __forceinline__ v8f wmma_bf16(v16bf a, v16bf b, v8f c) {
  return __builtin_amdgcn_wmma_f32_16x16x32_bf16(false, a, false, b,
                                                 (short)0, c, false, false);
}

// Fragment-order LDS slot for A element (m, k) of a 16x32 bf16 tile.
// Inverse of ISA 7.12.2 A layout: lane = ((k>>3)&1)*16 + m,
// pair p = ((k&16)>>2) + ((k&7)>>1), elem = k&1; 16 bf16 per lane.
__device__ __forceinline__ int a_slot(int m, int k) {
  const int hf = (k >> 3) & 1;
  const int p  = ((k & 16) >> 2) + ((k & 7) >> 1);
  return (((hf << 4) + m) << 4) + (p << 1) + (k & 1);
}
// Fragment-order LDS slot for B element (k, n) of a 32x16 bf16 tile.
// lane = ((k>>3)&1)*16 + n, pair p = ((k&16)>>2) + ((k&7)>>1), elem = k&1.
__device__ __forceinline__ int b_slot(int k, int n) {
  const int hf = (k >> 3) & 1;
  const int p  = ((k & 16) >> 2) + ((k & 7) >> 1);
  return (((hf << 4) + n) << 4) + (p << 1) + (k & 1);
}

__device__ __forceinline__ float snake_act(float r, float a) {
  const float s = sinf(a * r);
  return r + s * s / a;
}

// ------------------------- encoder conv (implicit GEMM) --------------------
// y[b,co,t] = snake(bias[co] + sum_{ci,tap} w[co,ci,tap] * x[b,ci,t*S-P+tap])
template <int KT, int STRIDE, int PAD>
__global__ __launch_bounds__(32)
void conv_enc_wmma(const float* __restrict__ x, const float* __restrict__ w,
                   const float* __restrict__ bias, const float* __restrict__ alpha,
                   float* __restrict__ y, int Cin, int Cout, int Lin, int Lout) {
  __shared__ __align__(16) __bf16 AsF[2][512];
  __shared__ __align__(16) __bf16 BsF[2][512];
  const int lane = threadIdx.x & 31;
  const int t0 = blockIdx.x * 32, c0 = blockIdx.y * 32, b = blockIdx.z;
  const int Ktot = Cin * KT;
  const float* xb = x + (size_t)b * Cin * Lin;
  const v8f vzero = {0.f, 0.f, 0.f, 0.f, 0.f, 0.f, 0.f, 0.f};
  v8f acc[2][2] = {{vzero, vzero}, {vzero, vzero}};

  for (int kk = 0; kk < Ktot; kk += 32) {
    if (kk + 32 < Ktot)
      __builtin_prefetch(w + (size_t)c0 * Ktot + kk + 32, 0, 1);
#pragma unroll
    for (int mi = 0; mi < 2; ++mi) {          // stage A tiles (weights)
#pragma unroll
      for (int pe = 0; pe < 8; ++pe) {
        const int q = lane * 8 + pe;          // pair id 0..255
        const int row = q >> 4;
        const int k0 = (q & 15) << 1;
        const int co = c0 + mi * 16 + row;
        const int kidx = kk + k0;
        const bool okr = (co < Cout);
        const bool ok0 = okr && (kidx < Ktot);
        const bool ok1 = okr && (kidx + 1 < Ktot);
        const size_t base = (size_t)(okr ? co : 0) * Ktot;
        float f0 = w[base + (ok0 ? kidx : 0)];
        float f1 = w[base + (ok1 ? kidx + 1 : 0)];
        f0 = ok0 ? f0 : 0.f;
        f1 = ok1 ? f1 : 0.f;
        v2bf pv = {to_bf(f0), to_bf(f1)};
        *(v2bf*)&AsF[mi][a_slot(row, k0)] = pv;
      }
    }
#pragma unroll
    for (int ni = 0; ni < 2; ++ni) {          // stage B tiles (im2col)
#pragma unroll
      for (int pe = 0; pe < 8; ++pe) {
        const int q = lane * 8 + pe;
        const int k0 = (q >> 4) << 1;
        const int nn = q & 15;
        const int t = t0 + ni * 16 + nn;
        float f[2];
#pragma unroll
        for (int ee = 0; ee < 2; ++ee) {
          const int kidx = kk + k0 + ee;
          const int ci = kidx / KT;           // KT is constexpr
          const int tap = kidx - ci * KT;
          const int ti = t * STRIDE - PAD + tap;
          const bool ok = (kidx < Ktot) && (ti >= 0) && (ti < Lin);
          const float lv = xb[(size_t)(ok ? ci : 0) * Lin + (ok ? ti : 0)];
          f[ee] = ok ? lv : 0.f;
        }
        v2bf pv = {to_bf(f[0]), to_bf(f[1])};
        *(v2bf*)&BsF[ni][b_slot(k0, nn)] = pv;
      }
    }
    __syncthreads();
    {
      const v16bf a0 = *(const v16bf*)&AsF[0][lane * 16];
      const v16bf a1 = *(const v16bf*)&AsF[1][lane * 16];
      const v16bf b0 = *(const v16bf*)&BsF[0][lane * 16];
      const v16bf b1 = *(const v16bf*)&BsF[1][lane * 16];
      acc[0][0] = wmma_bf16(a0, b0, acc[0][0]);
      acc[0][1] = wmma_bf16(a0, b1, acc[0][1]);
      acc[1][0] = wmma_bf16(a1, b0, acc[1][0]);
      acc[1][1] = wmma_bf16(a1, b1, acc[1][1]);
    }
    __syncthreads();
  }

  const int nn = lane & 15, hf = lane >> 4;
#pragma unroll
  for (int mi = 0; mi < 2; ++mi) {
#pragma unroll
    for (int vi = 0; vi < 8; ++vi) {
      const int c = c0 + mi * 16 + hf * 8 + vi;
      if (c < Cout) {
        const float bi = bias[c];
        const float a = alpha[c];
#pragma unroll
        for (int ni = 0; ni < 2; ++ni) {
          const int t = t0 + ni * 16 + nn;
          if (t < Lout)
            y[((size_t)b * Cout + c) * Lout + t] = snake_act(acc[mi][ni][vi] + bi, a);
        }
      }
    }
  }
}

// ------------------------ decoder transposed conv --------------------------
// y[b,co,t] = act(bias[co] + sum_{ci,j} w[ci,co,KT-1-j] * xd[t-(KT-1-PAD)+j])
template <int KT, int STRIDE, int PAD, int ACT>   // ACT: 0 snake, 1 tanh
__global__ __launch_bounds__(32)
void conv_dec_wmma(const float* __restrict__ x, const float* __restrict__ w,
                   const float* __restrict__ bias, const float* __restrict__ alpha,
                   float* __restrict__ y, int Cin, int Cout, int Lin, int Lout) {
  __shared__ __align__(16) __bf16 AsF[2][512];
  __shared__ __align__(16) __bf16 BsF[2][512];
  const int lane = threadIdx.x & 31;
  const int t0 = blockIdx.x * 32, c0 = blockIdx.y * 32, b = blockIdx.z;
  const int Ktot = Cin * KT;
  const int padT = KT - 1 - PAD;
  const int dilLen = (Lin - 1) * STRIDE + 1;
  const float* xb = x + (size_t)b * Cin * Lin;
  const v8f vzero = {0.f, 0.f, 0.f, 0.f, 0.f, 0.f, 0.f, 0.f};
  v8f acc[2][2] = {{vzero, vzero}, {vzero, vzero}};

  for (int kk = 0; kk < Ktot; kk += 32) {
#pragma unroll
    for (int mi = 0; mi < 2; ++mi) {          // stage A: flipped/transposed w
#pragma unroll
      for (int pe = 0; pe < 8; ++pe) {
        const int q = lane * 8 + pe;
        const int row = q >> 4;
        const int k0 = (q & 15) << 1;
        const int co = c0 + mi * 16 + row;
        float f[2];
#pragma unroll
        for (int ee = 0; ee < 2; ++ee) {
          const int kidx = kk + k0 + ee;
          const int ci = kidx / KT;
          const int j = kidx - ci * KT;
          const bool ok = (co < Cout) && (kidx < Ktot);
          const float lv = w[((size_t)(ok ? ci : 0) * Cout + (ok ? co : 0)) * KT +
                             (KT - 1 - j)];
          f[ee] = ok ? lv : 0.f;
        }
        v2bf pv = {to_bf(f[0]), to_bf(f[1])};
        *(v2bf*)&AsF[mi][a_slot(row, k0)] = pv;
      }
    }
#pragma unroll
    for (int ni = 0; ni < 2; ++ni) {          // stage B: dilated im2col
#pragma unroll
      for (int pe = 0; pe < 8; ++pe) {
        const int q = lane * 8 + pe;
        const int k0 = (q >> 4) << 1;
        const int nn = q & 15;
        const int t = t0 + ni * 16 + nn;
        float f[2];
#pragma unroll
        for (int ee = 0; ee < 2; ++ee) {
          const int kidx = kk + k0 + ee;
          const int ci = kidx / KT;
          const int j = kidx - ci * KT;
          const int md = t - padT + j;
          const bool ok = (kidx < Ktot) && (md >= 0) && (md < dilLen) &&
                          ((md % STRIDE) == 0);
          const int src = ok ? (md / STRIDE) : 0;
          const float lv = xb[(size_t)(ok ? ci : 0) * Lin + src];
          f[ee] = ok ? lv : 0.f;
        }
        v2bf pv = {to_bf(f[0]), to_bf(f[1])};
        *(v2bf*)&BsF[ni][b_slot(k0, nn)] = pv;
      }
    }
    __syncthreads();
    {
      const v16bf a0 = *(const v16bf*)&AsF[0][lane * 16];
      const v16bf a1 = *(const v16bf*)&AsF[1][lane * 16];
      const v16bf b0 = *(const v16bf*)&BsF[0][lane * 16];
      const v16bf b1 = *(const v16bf*)&BsF[1][lane * 16];
      acc[0][0] = wmma_bf16(a0, b0, acc[0][0]);
      acc[0][1] = wmma_bf16(a0, b1, acc[0][1]);
      acc[1][0] = wmma_bf16(a1, b0, acc[1][0]);
      acc[1][1] = wmma_bf16(a1, b1, acc[1][1]);
    }
    __syncthreads();
  }

  const int nn = lane & 15, hf = lane >> 4;
#pragma unroll
  for (int mi = 0; mi < 2; ++mi) {
#pragma unroll
    for (int vi = 0; vi < 8; ++vi) {
      const int c = c0 + mi * 16 + hf * 8 + vi;
      if (c < Cout) {
        const float bi = bias[c];
        const float a = (ACT == 0) ? alpha[c] : 1.f;
#pragma unroll
        for (int ni = 0; ni < 2; ++ni) {
          const int t = t0 + ni * 16 + nn;
          if (t < Lout) {
            float r = acc[mi][ni][vi] + bi;
            r = (ACT == 1) ? tanhf(r) : snake_act(r, a);
            y[((size_t)b * Cout + c) * Lout + t] = r;
          }
        }
      }
    }
  }
}

// ------------------------------ RVQ GEMMs ----------------------------------
// Shared staging for an unguarded row-major A (M x K) tile pair.
__device__ __forceinline__ void stage_a_dense(const float* __restrict__ A,
                                              int m0, int kk, int ldk,
                                              __bf16 (*AsF)[512], int lane) {
#pragma unroll
  for (int mi = 0; mi < 2; ++mi) {
#pragma unroll
    for (int pe = 0; pe < 8; ++pe) {
      const int q = lane * 8 + pe;
      const int row = q >> 4;
      const int k0 = (q & 15) << 1;
      const float* p = A + (size_t)(m0 + mi * 16 + row) * ldk + kk + k0;
      v2bf pv = {to_bf(p[0]), to_bf(p[1])};
      *(v2bf*)&AsF[mi][a_slot(row, k0)] = pv;
    }
  }
}

// zp[m,n] = pin_w[m,:] . residual[b(n), :, t(n)] + pin_b[m]   (M=128, K=512)
__global__ __launch_bounds__(32)
void rvq_pin(const float* __restrict__ res, const float* __restrict__ pw,
             const float* __restrict__ pb, float* __restrict__ zp) {
  __shared__ __align__(16) __bf16 AsF[2][512];
  __shared__ __align__(16) __bf16 BsF[2][512];
  const int lane = threadIdx.x & 31;
  const int n0 = blockIdx.x * 32, m0 = blockIdx.y * 32;
  const v8f vzero = {0.f, 0.f, 0.f, 0.f, 0.f, 0.f, 0.f, 0.f};
  v8f acc[2][2] = {{vzero, vzero}, {vzero, vzero}};

  for (int kk = 0; kk < DLAT; kk += 32) {
    stage_a_dense(pw, m0, kk, DLAT, AsF, lane);
#pragma unroll
    for (int ni = 0; ni < 2; ++ni) {
#pragma unroll
      for (int pe = 0; pe < 8; ++pe) {
        const int q = lane * 8 + pe;
        const int k0 = (q >> 4) << 1;
        const int nn = q & 15;
        const int n = n0 + ni * 16 + nn;
        const bool ok = (n < BT);
        const int ns = ok ? n : 0;
        const int b = ns / TLAT, t = ns - b * TLAT;
        const float* p = res + ((size_t)b * DLAT + kk + k0) * TLAT + t;
        const float f0 = ok ? p[0] : 0.f;
        const float f1 = ok ? p[TLAT] : 0.f;
        v2bf pv = {to_bf(f0), to_bf(f1)};
        *(v2bf*)&BsF[ni][b_slot(k0, nn)] = pv;
      }
    }
    __syncthreads();
    {
      const v16bf a0 = *(const v16bf*)&AsF[0][lane * 16];
      const v16bf a1 = *(const v16bf*)&AsF[1][lane * 16];
      const v16bf b0 = *(const v16bf*)&BsF[0][lane * 16];
      const v16bf b1 = *(const v16bf*)&BsF[1][lane * 16];
      acc[0][0] = wmma_bf16(a0, b0, acc[0][0]);
      acc[0][1] = wmma_bf16(a0, b1, acc[0][1]);
      acc[1][0] = wmma_bf16(a1, b0, acc[1][0]);
      acc[1][1] = wmma_bf16(a1, b1, acc[1][1]);
    }
    __syncthreads();
  }

  const int nn = lane & 15, hf = lane >> 4;
#pragma unroll
  for (int mi = 0; mi < 2; ++mi) {
#pragma unroll
    for (int vi = 0; vi < 8; ++vi) {
      const int m = m0 + mi * 16 + hf * 8 + vi;
      const float bi = pb[m];
#pragma unroll
      for (int ni = 0; ni < 2; ++ni) {
        const int n = n0 + ni * 16 + nn;
        if (n < BT) zp[(size_t)m * BT + n] = acc[mi][ni][vi] + bi;
      }
    }
  }
}

// sc[m,n] = ||cb[m]||^2 - 2 * cb[m,:] . zp[:,n]   (M=1024, K=128)
__global__ __launch_bounds__(32)
void rvq_score(const float* __restrict__ cb, const float* __restrict__ zp,
               const float* __restrict__ cbn, float* __restrict__ sc) {
  __shared__ __align__(16) __bf16 AsF[2][512];
  __shared__ __align__(16) __bf16 BsF[2][512];
  const int lane = threadIdx.x & 31;
  const int n0 = blockIdx.x * 32, m0 = blockIdx.y * 32;
  const v8f vzero = {0.f, 0.f, 0.f, 0.f, 0.f, 0.f, 0.f, 0.f};
  v8f acc[2][2] = {{vzero, vzero}, {vzero, vzero}};

  for (int kk = 0; kk < DPROJ; kk += 32) {
    stage_a_dense(cb, m0, kk, DPROJ, AsF, lane);
#pragma unroll
    for (int ni = 0; ni < 2; ++ni) {
#pragma unroll
      for (int pe = 0; pe < 8; ++pe) {
        const int q = lane * 8 + pe;
        const int k0 = (q >> 4) << 1;
        const int nn = q & 15;
        const int n = n0 + ni * 16 + nn;
        const bool ok = (n < BT);
        const float* p = zp + (size_t)(kk + k0) * BT + (ok ? n : 0);
        const float f0 = ok ? p[0] : 0.f;
        const float f1 = ok ? p[BT] : 0.f;
        v2bf pv = {to_bf(f0), to_bf(f1)};
        *(v2bf*)&BsF[ni][b_slot(k0, nn)] = pv;
      }
    }
    __syncthreads();
    {
      const v16bf a0 = *(const v16bf*)&AsF[0][lane * 16];
      const v16bf a1 = *(const v16bf*)&AsF[1][lane * 16];
      const v16bf b0 = *(const v16bf*)&BsF[0][lane * 16];
      const v16bf b1 = *(const v16bf*)&BsF[1][lane * 16];
      acc[0][0] = wmma_bf16(a0, b0, acc[0][0]);
      acc[0][1] = wmma_bf16(a0, b1, acc[0][1]);
      acc[1][0] = wmma_bf16(a1, b0, acc[1][0]);
      acc[1][1] = wmma_bf16(a1, b1, acc[1][1]);
    }
    __syncthreads();
  }

  const int nn = lane & 15, hf = lane >> 4;
#pragma unroll
  for (int mi = 0; mi < 2; ++mi) {
#pragma unroll
    for (int vi = 0; vi < 8; ++vi) {
      const int m = m0 + mi * 16 + hf * 8 + vi;
      const float cn = cbn[m];
#pragma unroll
      for (int ni = 0; ni < 2; ++ni) {
        const int n = n0 + ni * 16 + nn;
        if (n < BT) sc[(size_t)m * BT + n] = cn - 2.f * acc[mi][ni][vi];
      }
    }
  }
}

// residual[b(n), m, t(n)] -= pout_w[m,:] . zq[:,n] + pout_b[m]  (M=512, K=128)
__global__ __launch_bounds__(32)
void rvq_pout(const float* __restrict__ zq, const float* __restrict__ pw,
              const float* __restrict__ pb, float* __restrict__ res) {
  __shared__ __align__(16) __bf16 AsF[2][512];
  __shared__ __align__(16) __bf16 BsF[2][512];
  const int lane = threadIdx.x & 31;
  const int n0 = blockIdx.x * 32, m0 = blockIdx.y * 32;
  const v8f vzero = {0.f, 0.f, 0.f, 0.f, 0.f, 0.f, 0.f, 0.f};
  v8f acc[2][2] = {{vzero, vzero}, {vzero, vzero}};

  for (int kk = 0; kk < DPROJ; kk += 32) {
    stage_a_dense(pw, m0, kk, DPROJ, AsF, lane);
#pragma unroll
    for (int ni = 0; ni < 2; ++ni) {
#pragma unroll
      for (int pe = 0; pe < 8; ++pe) {
        const int q = lane * 8 + pe;
        const int k0 = (q >> 4) << 1;
        const int nn = q & 15;
        const int n = n0 + ni * 16 + nn;
        const bool ok = (n < BT);
        const float* p = zq + (size_t)(kk + k0) * BT + (ok ? n : 0);
        const float f0 = ok ? p[0] : 0.f;
        const float f1 = ok ? p[BT] : 0.f;
        v2bf pv = {to_bf(f0), to_bf(f1)};
        *(v2bf*)&BsF[ni][b_slot(k0, nn)] = pv;
      }
    }
    __syncthreads();
    {
      const v16bf a0 = *(const v16bf*)&AsF[0][lane * 16];
      const v16bf a1 = *(const v16bf*)&AsF[1][lane * 16];
      const v16bf b0 = *(const v16bf*)&BsF[0][lane * 16];
      const v16bf b1 = *(const v16bf*)&BsF[1][lane * 16];
      acc[0][0] = wmma_bf16(a0, b0, acc[0][0]);
      acc[0][1] = wmma_bf16(a0, b1, acc[0][1]);
      acc[1][0] = wmma_bf16(a1, b0, acc[1][0]);
      acc[1][1] = wmma_bf16(a1, b1, acc[1][1]);
    }
    __syncthreads();
  }

  const int nn = lane & 15, hf = lane >> 4;
#pragma unroll
  for (int mi = 0; mi < 2; ++mi) {
#pragma unroll
    for (int vi = 0; vi < 8; ++vi) {
      const int m = m0 + mi * 16 + hf * 8 + vi;
      const float bi = pb[m];
#pragma unroll
      for (int ni = 0; ni < 2; ++ni) {
        const int n = n0 + ni * 16 + nn;
        if (n < BT) {
          const int b = n / TLAT, t = n - b * TLAT;
          res[((size_t)b * DLAT + m) * TLAT + t] -= acc[mi][ni][vi] + bi;
        }
      }
    }
  }
}

// -------------------------- RVQ scalar kernels -----------------------------
__global__ void cb_norms(const float* __restrict__ cbs, float* __restrict__ cbn) {
  const int i = blockIdx.x * blockDim.x + threadIdx.x;
  if (i < NBOOKS * NCODES) {
    const float* p = cbs + (size_t)i * DPROJ;
    float s = 0.f;
    for (int c = 0; c < DPROJ; ++c) { const float v = p[c]; s += v * v; }
    cbn[i] = s;
  }
}

__global__ __launch_bounds__(256)
void rvq_argmin(const float* __restrict__ sc, int* __restrict__ codes, int book) {
  __shared__ float sv[256];
  __shared__ int   si[256];
  const int bt = blockIdx.x, tid = threadIdx.x;
  float best = 3.4e38f;
  int bidx = 0;
  for (int n = tid; n < NCODES; n += 256) {
    const float v = sc[(size_t)n * BT + bt];
    if (v < best) { best = v; bidx = n; }
  }
  sv[tid] = best; si[tid] = bidx;
  __syncthreads();
  for (int off = 128; off > 0; off >>= 1) {
    if (tid < off) {
      const float v2 = sv[tid + off]; const int i2 = si[tid + off];
      if (v2 < sv[tid] || (v2 == sv[tid] && i2 < si[tid])) { sv[tid] = v2; si[tid] = i2; }
    }
    __syncthreads();
  }
  if (tid == 0) {
    const int b = bt / TLAT, t = bt - b * TLAT;
    codes[(size_t)b * (NBOOKS * TLAT) + book * TLAT + t] = si[0];
  }
}

__global__ __launch_bounds__(128)
void rvq_gather(const float* __restrict__ cb, const int* __restrict__ codes,
                int book, const float* __restrict__ zp, float* __restrict__ zq,
                float* __restrict__ loss) {
  __shared__ float red[128];
  const int bt = blockIdx.x, c = threadIdx.x;
  const int b = bt / TLAT, t = bt - b * TLAT;
  const int idx = codes[(size_t)b * (NBOOKS * TLAT) + book * TLAT + t];
  const float zv = cb[(size_t)idx * DPROJ + c];
  const float d = zv - zp[(size_t)c * BT + bt];
  zq[(size_t)c * BT + bt] = zv;
  red[c] = d * d;
  __syncthreads();
  for (int off = 64; off > 0; off >>= 1) {
    if (c < off) red[c] += red[c + off];
    __syncthreads();
  }
  if (c == 0) atomicAdd(loss, red[0]);
}

// ------------------------------ misc ---------------------------------------
__global__ void init_loss(float* l) { *l = 0.f; }

__global__ void sub_z(const float* __restrict__ z, const float* __restrict__ r,
                      float* __restrict__ o, int n) {
  const int i = blockIdx.x * blockDim.x + threadIdx.x;
  if (i < n) o[i] = z[i] - r[i];
}

__global__ void write_loss(const float* __restrict__ acc, float* __restrict__ out) {
  // commit == cbloss in value; total = (1 + 0.25) * sum_sq / (64*249*128)
  out[0] = 1.25f * acc[0] / 2039808.0f;
}

// ------------------------------ host side ----------------------------------
// workspace layout (float offsets)
static constexpr size_t OFF_A   = 0;          // 61,440,000  (enc0 out / dec2 out)
static constexpr size_t OFF_B   = 61440000;   // 24,567,808  (enc1 out / dec1 out)
static constexpr size_t OFF_C   = 86007808;   // 12,271,616  (enc2 out / dec0 out)
static constexpr size_t OFF_Z   = 98279424;   //  8,159,232  (latent z)
static constexpr size_t OFF_R   = 106438656;  //  8,159,232  (residual)
static constexpr size_t OFF_QF  = 114597888;  //  8,159,232  (z_q = z - residual)
static constexpr size_t OFF_ZP  = 122757120;  //  2,039,808  (zp [128][BT])
static constexpr size_t OFF_ZQ  = 124796928;  //  2,039,808  (zq [128][BT])
static constexpr size_t OFF_SC  = 126836736;  // 16,318,464  (scores [1024][BT])
static constexpr size_t OFF_CBN = 143155200;  //      8,192
static constexpr size_t OFF_LS  = 143163392;  //          1

extern "C" void kernel_launch(void* const* d_in, const int* in_sizes, int n_in,
                              void* d_out, int out_size, void* d_ws, size_t ws_size,
                              hipStream_t stream) {
  (void)in_sizes; (void)n_in; (void)out_size; (void)ws_size;
  const float* audio = (const float*)d_in[0];
  const float* ew[4] = {(const float*)d_in[1], (const float*)d_in[4],
                        (const float*)d_in[7], (const float*)d_in[10]};
  const float* eb[4] = {(const float*)d_in[2], (const float*)d_in[5],
                        (const float*)d_in[8], (const float*)d_in[11]};
  const float* ea[4] = {(const float*)d_in[3], (const float*)d_in[6],
                        (const float*)d_in[9], (const float*)d_in[12]};
  const float* dw[4] = {(const float*)d_in[13], (const float*)d_in[16],
                        (const float*)d_in[19], (const float*)d_in[22]};
  const float* db[4] = {(const float*)d_in[14], (const float*)d_in[17],
                        (const float*)d_in[20], (const float*)d_in[23]};
  const float* da[3] = {(const float*)d_in[15], (const float*)d_in[18],
                        (const float*)d_in[21]};
  const float* pin_w  = (const float*)d_in[24];
  const float* pin_b  = (const float*)d_in[25];
  const float* pout_w = (const float*)d_in[26];
  const float* pout_b = (const float*)d_in[27];
  const float* cbs    = (const float*)d_in[28];

  float* ws  = (float*)d_ws;
  float* bA  = ws + OFF_A;
  float* bB  = ws + OFF_B;
  float* bC  = ws + OFF_C;
  float* z   = ws + OFF_Z;
  float* res = ws + OFF_R;
  float* zqf = ws + OFF_QF;
  float* zp  = ws + OFF_ZP;
  float* zq  = ws + OFF_ZQ;
  float* sc  = ws + OFF_SC;
  float* cbn = ws + OFF_CBN;
  float* ls  = ws + OFF_LS;

  float* out_audio = (float*)d_out;                      // 64*119688
  int*   out_codes = (int*)((float*)d_out + 7660032);    // 64*8*249
  float* out_loss  = (float*)d_out + 7787520;            // 1

  // ---------------- encoder ----------------
  conv_enc_wmma<16, 8, 4><<<dim3(469, 2, BATCH), 32, 0, stream>>>(
      audio, ew[0], eb[0], ea[0], bA, 1, 64, 120000, 15000);
  conv_enc_wmma<10, 5, 2><<<dim3(94, 4, BATCH), 32, 0, stream>>>(
      bA, ew[1], eb[1], ea[1], bB, 64, 128, 15000, 2999);
  conv_enc_wmma<8, 4, 2><<<dim3(24, 8, BATCH), 32, 0, stream>>>(
      bB, ew[2], eb[2], ea[2], bC, 128, 256, 2999, 749);
  conv_enc_wmma<6, 3, 1><<<dim3(8, 16, BATCH), 32, 0, stream>>>(
      bC, ew[3], eb[3], ea[3], z, 256, 512, 749, 249);

  // ---------------- RVQ ----------------
  cb_norms<<<(NBOOKS * NCODES + 255) / 256, 256, 0, stream>>>(cbs, cbn);
  init_loss<<<1, 1, 0, stream>>>(ls);
  hipMemcpyAsync(res, z, (size_t)BATCH * DLAT * TLAT * sizeof(float),
                 hipMemcpyDeviceToDevice, stream);

  for (int book = 0; book < NBOOKS; ++book) {
    const float* cb = cbs + (size_t)book * NCODES * DPROJ;
    rvq_pin  <<<dim3(498,  4, 1), 32, 0, stream>>>(res, pin_w, pin_b, zp);
    rvq_score<<<dim3(498, 32, 1), 32, 0, stream>>>(cb, zp, cbn + book * NCODES, sc);
    rvq_argmin<<<BT, 256, 0, stream>>>(sc, out_codes, book);
    rvq_gather<<<BT, 128, 0, stream>>>(cb, out_codes, book, zp, zq, ls);
    rvq_pout <<<dim3(498, 16, 1), 32, 0, stream>>>(zq, pout_w, pout_b, res);
  }

  sub_z<<<(BATCH * DLAT * TLAT + 255) / 256, 256, 0, stream>>>(
      z, res, zqf, BATCH * DLAT * TLAT);

  // ---------------- decoder ----------------
  conv_dec_wmma<6, 3, 1, 0><<<dim3(24, 8, BATCH), 32, 0, stream>>>(
      zqf, dw[0], db[0], da[0], bC, 512, 256, 249, 748);
  conv_dec_wmma<8, 4, 2, 0><<<dim3(94, 4, BATCH), 32, 0, stream>>>(
      bC, dw[1], db[1], da[1], bB, 256, 128, 748, 2992);
  conv_dec_wmma<10, 5, 2, 0><<<dim3(468, 2, BATCH), 32, 0, stream>>>(
      bB, dw[2], db[2], da[2], bA, 128, 64, 2992, 14961);
  conv_dec_wmma<16, 8, 4, 1><<<dim3(3741, 1, BATCH), 32, 0, stream>>>(
      bA, dw[3], db[3], nullptr, out_audio, 64, 1, 14961, 119688);

  write_loss<<<1, 1, 0, stream>>>(ls, out_loss);
}
